// WorldModel_83915071030080
// MI455X (gfx1250) — matrix operand order repaired and checked by reading, not compiled, not run on hardware
//
#include <hip/hip_runtime.h>
#include <hip/hip_bf16.h>

// ---------------------------------------------------------------------------
// CDNA5 (gfx1250) bf16 WMMA RSSM rollout.
// D = A(16x32 bf16) x B(32x16 bf16) + C(16x16 f32) via v_wmma_f32_16x16x32_bf16
// ---------------------------------------------------------------------------

typedef __bf16 v16bf __attribute__((ext_vector_type(16)));
typedef float  v8f   __attribute__((ext_vector_type(8)));

#define Bsz   512
#define Hdim  200
#define HPAD  224     // 7 K-tiles of 32
#define LCdim 1024
#define Edim  1536
#define OUTC  3272    // 200 + 1024 + 1024 + 1024
#define Tlen  50

__host__ __device__ static inline unsigned rotl32(unsigned x, int r) {
    return (x << r) | (x >> (32 - r));
}
// Threefry-2x32, 20 rounds (JAX PRNG core)
__host__ __device__ static inline void tf2x32(unsigned k0, unsigned k1,
                                              unsigned c0, unsigned c1,
                                              unsigned& o0, unsigned& o1) {
    unsigned ks2 = 0x1BD11BDAu ^ k0 ^ k1;
    unsigned x0 = c0 + k0, x1 = c1 + k1;
    const int ra[4] = {13, 15, 26, 6};
    const int rb[4] = {17, 29, 16, 24};
#define TFR4(R) { x0 += x1; x1 = rotl32(x1, R[0]) ^ x0; \
                  x0 += x1; x1 = rotl32(x1, R[1]) ^ x0; \
                  x0 += x1; x1 = rotl32(x1, R[2]) ^ x0; \
                  x0 += x1; x1 = rotl32(x1, R[3]) ^ x0; }
    TFR4(ra); x0 += k1;  x1 += ks2 + 1u;
    TFR4(rb); x0 += ks2; x1 += k0 + 2u;
    TFR4(ra); x0 += k0;  x1 += k1 + 3u;
    TFR4(rb); x0 += k1;  x1 += ks2 + 4u;
    TFR4(ra); x0 += ks2; x1 += k0 + 5u;
#undef TFR4
    o0 = x0; o1 = x1;
}

__device__ __forceinline__ unsigned short f2bf(float f) {
    unsigned u = __float_as_uint(f);
    unsigned r = u + 0x7FFFu + ((u >> 16) & 1u);   // round-to-nearest-even
    return (unsigned short)(r >> 16);
}

union Frag { v16bf v; uint4 q[2]; };

// A fragment: 16x32 bf16 tile, row-major (stride ld halves).
// ISA layout: lanes 0-15 (rows M=0..15) hold K {0..7,16..23}; lanes 16-31 K {8..15,24..31}.
__device__ __forceinline__ v16bf load_frag_a(const unsigned short* tile, int ld) {
    int lane = threadIdx.x & 31;
    int h = lane >> 4, r = lane & 15;
    const unsigned short* p = tile + r * ld + h * 8;
    Frag f;
    f.q[0] = *(const uint4*)(p);        // K = h*8 .. h*8+7
    f.q[1] = *(const uint4*)(p + 16);   // K = h*8+16 .. h*8+23
    return f.v;
}
// B fragment: 32x16 bf16; W stored TRANSPOSED: value B[k][n] at wt[n*ld + k].
// lanes 0-15 -> N=lane (K {0..7,16..23}), lanes 16-31 -> N=lane-16 (K {8..15,24..31}); 2 K per VGPR.
__device__ __forceinline__ v16bf load_frag_b(const unsigned short* wt, int ld) {
    int lane = threadIdx.x & 31;
    int kh = lane >> 4, n = lane & 15;
    const unsigned short* p = wt + n * ld + kh * 8;
    Frag f;
    f.q[0] = *(const uint4*)(p);
    f.q[1] = *(const uint4*)(p + 16);
    return f.v;
}

// Stage one 16x32 bf16 A-tile into LDS (128-thread block: each thread 8 bytes)
__device__ __forceinline__ void stage_a(unsigned short* sA, const unsigned short* A,
                                        int lda, int m0, int kt) {
    int t = threadIdx.x;
    int row  = t >> 3;
    int colu = (t & 7) * 4;
    *(uint2*)(sA + row * 32 + colu) =
        *(const uint2*)(A + (size_t)(m0 + row) * lda + kt * 32 + colu);
}

// ---------------------------------------------------------------------------
// Plain GEMM: out[m, col] = sum_k A[m,k]*W[k,col] + bias[col]
// Block = 128 threads (4 waves), 16 rows x 4 N-tiles. Grid (M/16, ceil(ntiles/4)).
// ---------------------------------------------------------------------------
__global__ void __launch_bounds__(128) gemm_plain_k(
    const unsigned short* __restrict__ A0, int lda0,
    const unsigned short* __restrict__ W0, int ldw0, int kt0,
    const float* __restrict__ bias,
    float* __restrict__ out, int ldc, int Nvalid, int ntiles)
{
    __shared__ unsigned short sA[16 * 32];
    int m0   = blockIdx.x * 16;
    int wave = threadIdx.x >> 5;
    int lane = threadIdx.x & 31;
    int ntile = blockIdx.y * 4 + wave;
    bool active = (ntile < ntiles);
    v8f acc = {0.f, 0.f, 0.f, 0.f, 0.f, 0.f, 0.f, 0.f};

    for (int kt = 0; kt < kt0; ++kt) {
        __syncthreads();
        stage_a(sA, A0, lda0, m0, kt);
        __syncthreads();
        if (active) {
            if (kt + 1 < kt0)
                __builtin_prefetch(W0 + (size_t)ntile * 16 * ldw0 + (kt + 1) * 32, 0, 1);
            v16bf a = load_frag_a(sA, 32);
            v16bf b = load_frag_b(W0 + (size_t)ntile * 16 * ldw0 + kt * 32, ldw0);
            acc = __builtin_amdgcn_wmma_f32_16x16x32_bf16(false, a, false, b,
                                                          (short)0, acc, false, false);
        }
    }
    if (active) {
        int h = lane >> 4, n = lane & 15;
        int col = ntile * 16 + n;
        if (col < Nvalid) {
            float bv = bias ? bias[col] : 0.f;
#pragma unroll
            for (int r = 0; r < 8; ++r)
                out[(size_t)(m0 + r + 8 * h) * ldc + col] = acc[r] + bv;
        }
    }
}

// ---------------------------------------------------------------------------
// GEMM (up to 2 K-source pairs) + bias + optional fp32 tail-K + LayerNorm(row)
// + ELU, bf16 output. One block owns a full 16-row x Nvalid stripe so LN stats
// are computed in-block (wave shuffles + LDS cross-wave reduce).
// Block = 128 threads; grid = M/16; ntiles <= 16.
// ---------------------------------------------------------------------------
__global__ void __launch_bounds__(128) gemm_ln_elu_k(
    const unsigned short* __restrict__ A0, int lda0,
    const unsigned short* __restrict__ W0, int ldw0, int kt0,
    const unsigned short* __restrict__ A1, int lda1,
    const unsigned short* __restrict__ W1, int ldw1, int kt1,
    const float* __restrict__ bias, const float* __restrict__ lng,
    const float* __restrict__ lnb,
    const float* __restrict__ tailA, int tailLdA,
    const float* __restrict__ tailW, int tailLdW, int tailK,
    unsigned short* __restrict__ outbf, int ldout, int ntiles, int Nvalid)
{
    __shared__ unsigned short sA[16 * 32];
    __shared__ float red[2][4][16];
    __shared__ float stats[2][16];

    int m0   = blockIdx.x * 16;
    int wave = threadIdx.x >> 5;
    int lane = threadIdx.x & 31;
    int h = lane >> 4, n = lane & 15;

    v8f acc[4];
    v8f z = {0.f, 0.f, 0.f, 0.f, 0.f, 0.f, 0.f, 0.f};
#pragma unroll
    for (int i = 0; i < 4; ++i) acc[i] = z;

    for (int pair = 0; pair < 2; ++pair) {
        const unsigned short* A = pair ? A1 : A0;
        const unsigned short* W = pair ? W1 : W0;
        int lda = pair ? lda1 : lda0;
        int ldw = pair ? ldw1 : ldw0;
        int kt  = pair ? kt1  : kt0;
        if (!A) continue;
        for (int k = 0; k < kt; ++k) {
            __syncthreads();
            stage_a(sA, A, lda, m0, k);
            __syncthreads();
            v16bf af = load_frag_a(sA, 32);
            int cnt = 0;
            for (int tt = wave; tt < ntiles; tt += 4, ++cnt) {
                v16bf bfr = load_frag_b(W + (size_t)tt * 16 * ldw + k * 32, ldw);
                acc[cnt] = __builtin_amdgcn_wmma_f32_16x16x32_bf16(
                    false, af, false, bfr, (short)0, acc[cnt], false, false);
            }
        }
    }

    // bias + fp32 tail, then per-row mean/var partials
    float val[4][8];
    float ps[8], pq[8];
#pragma unroll
    for (int r = 0; r < 8; ++r) { ps[r] = 0.f; pq[r] = 0.f; }
    {
        int cnt = 0;
        for (int tt = wave; tt < ntiles; tt += 4, ++cnt) {
            int col = tt * 16 + n;
            bool ok = (col < Nvalid);
            float bv = (ok && bias) ? bias[col] : 0.f;
#pragma unroll
            for (int r = 0; r < 8; ++r) {
                float v = acc[cnt][r] + bv;
                if (ok && tailK > 0) {
                    int m = m0 + r + 8 * h;
                    for (int a = 0; a < tailK; ++a)
                        v += tailA[(size_t)m * tailLdA + a] * tailW[(size_t)a * tailLdW + col];
                }
                if (!ok) v = 0.f;
                val[cnt][r] = v;
                ps[r] += v;
                pq[r] += v * v;
            }
        }
    }
    // reduce over the 16 lanes of each half-wave (same set of 8 rows)
#pragma unroll
    for (int off = 1; off < 16; off <<= 1) {
#pragma unroll
        for (int r = 0; r < 8; ++r) {
            ps[r] += __shfl_xor(ps[r], off, 32);
            pq[r] += __shfl_xor(pq[r], off, 32);
        }
    }
    if (n == 0) {
#pragma unroll
        for (int r = 0; r < 8; ++r) {
            int row = r + 8 * h;
            red[0][wave][row] = ps[r];
            red[1][wave][row] = pq[r];
        }
    }
    __syncthreads();
    if (threadIdx.x < 16) {
        float s = 0.f, q = 0.f;
#pragma unroll
        for (int w = 0; w < 4; ++w) { s += red[0][w][threadIdx.x]; q += red[1][w][threadIdx.x]; }
        float mean = s / (float)Nvalid;
        float var  = q / (float)Nvalid - mean * mean;
        stats[0][threadIdx.x] = mean;
        stats[1][threadIdx.x] = rsqrtf(var + 1e-5f);
    }
    __syncthreads();

    {
        int cnt = 0;
        for (int tt = wave; tt < ntiles; tt += 4, ++cnt) {
            int col = tt * 16 + n;
            if (col < Nvalid) {
                float gg = lng[col], bb = lnb[col];
#pragma unroll
                for (int r = 0; r < 8; ++r) {
                    int m = r + 8 * h;
                    float y = (val[cnt][r] - stats[0][m]) * stats[1][m] * gg + bb;
                    y = (y > 0.f) ? y : expm1f(y);   // ELU
                    outbf[(size_t)(m0 + m) * ldout + col] = f2bf(y);
                }
            }
        }
    }
}

// ---------------------------------------------------------------------------
// GRU gates: h' = (1-z)*tanh(inn + r*hn) + z*h
// ---------------------------------------------------------------------------
__global__ void __launch_bounds__(256) gru_gate_k(
    const float* __restrict__ gi, const float* __restrict__ gh,
    float* __restrict__ hbuf, unsigned short* __restrict__ h_bf,
    float* __restrict__ out_h)
{
    int idx = blockIdx.x * 256 + threadIdx.x;
    if (idx >= Bsz * Hdim) return;
    int b = idx / Hdim, n = idx - b * Hdim;
    const float* gib = gi + (size_t)b * 600;
    const float* ghb = gh + (size_t)b * 600;
    float r  = 1.f / (1.f + expf(-(gib[n] + ghb[n])));
    float zz = 1.f / (1.f + expf(-(gib[n + 200] + ghb[n + 200])));
    float nn = tanhf(gib[n + 400] + r * ghb[n + 400]);
    float hv = (1.f - zz) * nn + zz * hbuf[idx];
    hbuf[idx] = hv;
    h_bf[b * HPAD + n] = f2bf(hv);
    out_h[(size_t)b * OUTC + n] = hv;
}

// ---------------------------------------------------------------------------
// Softmax over 32 classes + Gumbel-max categorical sample (threefry) +
// straight-through value (samp + p - p). One block per batch row, warp per latent.
// ---------------------------------------------------------------------------
__global__ void __launch_bounds__(256) sample_k(
    const float* __restrict__ logits, float* __restrict__ s_out,
    unsigned short* __restrict__ s_bf, unsigned k0, unsigned k1)
{
    int b = blockIdx.x;
    int w = threadIdx.x >> 5, c = threadIdx.x & 31;
#pragma unroll
    for (int i = 0; i < 4; ++i) {
        int l = w + 8 * i;
        float logit = logits[(size_t)b * OUTC + l * 32 + c];
        float mx = logit;
        for (int off = 16; off; off >>= 1) mx = fmaxf(mx, __shfl_xor(mx, off, 32));
        float e = expf(logit - mx);
        float se = e;
        for (int off = 16; off; off >>= 1) se += __shfl_xor(se, off, 32);
        float p = e / se;

        unsigned flat = (unsigned)((b * 32 + l) * 32 + c);
        unsigned r0, r1;
        tf2x32(k0, k1, 0u, flat, r0, r1);
        float u = ((float)(r0 >> 8) + 0.5f) * (1.0f / 16777216.0f);
        float key = logit - logf(-logf(u));   // Gumbel-max on logits

        float best = key; int bi = c;
        for (int off = 16; off; off >>= 1) {
            float ov = __shfl_xor(best, off, 32);
            int   oi = __shfl_xor(bi,   off, 32);
            if (ov > best || (ov == best && oi < bi)) { best = ov; bi = oi; }
        }
        float one = (c == bi) ? 1.f : 0.f;
        float sv = (one + p) - p;             // straight-through forward value
        s_out[(size_t)b * OUTC + l * 32 + c] = sv;
        s_bf[b * LCdim + l * 32 + c] = f2bf(sv);
    }
}

// --------------------------- prep / utility kernels ------------------------
// dst[n*Kpad + k] = bf16(src[k*srcLd + n]) with zero padding (transpose+pad)
__global__ void prep_wt_k(const float* __restrict__ src, int srcLd, int K, int N,
                          unsigned short* __restrict__ dst, int Kpad, int total)
{
    int idx = blockIdx.x * 256 + threadIdx.x;
    if (idx >= total) return;
    int n = idx / Kpad, k = idx - n * Kpad;
    float v = (n < N && k < K) ? src[(size_t)k * srcLd + n] : 0.f;
    dst[idx] = f2bf(v);
}
__global__ void init_bf_k(const float* __restrict__ src, unsigned short* __restrict__ dst,
                          int cols, int ldd, int total)
{
    int idx = blockIdx.x * 256 + threadIdx.x;
    if (idx >= total) return;
    int r = idx / ldd, c = idx - r * ldd;
    dst[idx] = (c < cols) ? f2bf(src[(size_t)r * cols + c]) : (unsigned short)0;
}
__global__ void conv_bf_k(const float* __restrict__ src, unsigned short* __restrict__ dst, int n)
{
    int idx = blockIdx.x * 256 + threadIdx.x;
    if (idx < n) dst[idx] = f2bf(src[idx]);
}
__global__ void copy_f_k(const float* __restrict__ src, float* __restrict__ dst, int n)
{
    int idx = blockIdx.x * 256 + threadIdx.x;
    if (idx < n) dst[idx] = src[idx];
}
__global__ void zero_us_k(unsigned short* __restrict__ dst, int n)
{
    int idx = blockIdx.x * 256 + threadIdx.x;
    if (idx < n) dst[idx] = 0;
}

// ---------------------------------------------------------------------------
extern "C" void kernel_launch(void* const* d_in, const int* in_sizes, int n_in,
                              void* d_out, int out_size, void* d_ws, size_t ws_size,
                              hipStream_t stream) {
    (void)in_sizes; (void)n_in; (void)out_size; (void)ws_size;
    const float* embed      = (const float*)d_in[0];
    const float* action     = (const float*)d_in[1];
    const float* determ0    = (const float*)d_in[2];
    const float* stoch0     = (const float*)d_in[3];
    const float* pre_W      = (const float*)d_in[4];
    const float* pre_b      = (const float*)d_in[5];
    const float* pre_ln_g   = (const float*)d_in[6];
    const float* pre_ln_b   = (const float*)d_in[7];
    const float* gru_Wih    = (const float*)d_in[8];
    const float* gru_Whh    = (const float*)d_in[9];
    const float* gru_bih    = (const float*)d_in[10];
    const float* gru_bhh    = (const float*)d_in[11];
    const float* ens_W1     = (const float*)d_in[12];
    const float* ens_b1     = (const float*)d_in[13];
    const float* ens_ln_g   = (const float*)d_in[14];
    const float* ens_ln_b   = (const float*)d_in[15];
    const float* ens_W2     = (const float*)d_in[16];
    const float* ens_b2     = (const float*)d_in[17];
    const float* post_W1    = (const float*)d_in[18];
    const float* post_b1    = (const float*)d_in[19];
    const float* post_ln_g  = (const float*)d_in[20];
    const float* post_ln_b  = (const float*)d_in[21];
    const float* post_W2    = (const float*)d_in[22];
    const float* post_b2    = (const float*)d_in[23];
    float* out = (float*)d_out;

    // -------- workspace bump allocator (256B aligned slices) --------
    char* wp = (char*)d_ws;
    auto alloc = [&](size_t bytes) -> void* {
        void* p = (void*)wp;
        wp += (bytes + 255) & ~(size_t)255;
        return p;
    };
    unsigned short* preWt    = (unsigned short*)alloc((size_t)208 * 1024 * 2);
    unsigned short* WihT     = (unsigned short*)alloc((size_t)608 * HPAD * 2);
    unsigned short* WhhT     = (unsigned short*)alloc((size_t)608 * HPAD * 2);
    unsigned short* ensW1T   = (unsigned short*)alloc((size_t)5 * 208 * HPAD * 2);
    unsigned short* ensW2T   = (unsigned short*)alloc((size_t)5 * 1024 * HPAD * 2);
    unsigned short* postW1hT = (unsigned short*)alloc((size_t)208 * HPAD * 2);
    unsigned short* postW1eT = (unsigned short*)alloc((size_t)208 * Edim * 2);
    unsigned short* postW2T  = (unsigned short*)alloc((size_t)1024 * HPAD * 2);
    unsigned short* s_bf     = (unsigned short*)alloc((size_t)Bsz * LCdim * 2);
    unsigned short* x_bf     = (unsigned short*)alloc((size_t)Bsz * HPAD * 2);
    unsigned short* h_bf     = (unsigned short*)alloc((size_t)Bsz * HPAD * 2);
    unsigned short* eh_bf    = (unsigned short*)alloc((size_t)Bsz * HPAD * 2);
    unsigned short* ph_bf    = (unsigned short*)alloc((size_t)Bsz * HPAD * 2);
    unsigned short* embT     = (unsigned short*)alloc((size_t)Bsz * Edim * 2);
    float* gi    = (float*)alloc((size_t)Bsz * 600 * 4);
    float* gh    = (float*)alloc((size_t)Bsz * 600 * 4);
    float* h_f32 = (float*)alloc((size_t)Bsz * Hdim * 4);

    auto blks = [](int n) { return (n + 255) / 256; };

    // -------- one-time prep: transpose+pad weights to bf16, init state ------
    prep_wt_k<<<blks(208 * 1024), 256, 0, stream>>>(pre_W, 200, 1024, 200, preWt, 1024, 208 * 1024);
    prep_wt_k<<<blks(608 * HPAD), 256, 0, stream>>>(gru_Wih, 600, 200, 600, WihT, HPAD, 608 * HPAD);
    prep_wt_k<<<blks(608 * HPAD), 256, 0, stream>>>(gru_Whh, 600, 200, 600, WhhT, HPAD, 608 * HPAD);
    for (int e = 0; e < 5; ++e) {
        prep_wt_k<<<blks(208 * HPAD), 256, 0, stream>>>(
            ens_W1 + (size_t)e * 200 * 200, 200, 200, 200,
            ensW1T + (size_t)e * 208 * HPAD, HPAD, 208 * HPAD);
        prep_wt_k<<<blks(1024 * HPAD), 256, 0, stream>>>(
            ens_W2 + (size_t)e * 200 * 1024, 1024, 200, 1024,
            ensW2T + (size_t)e * 1024 * HPAD, HPAD, 1024 * HPAD);
    }
    prep_wt_k<<<blks(208 * HPAD), 256, 0, stream>>>(post_W1, 200, 200, 200, postW1hT, HPAD, 208 * HPAD);
    prep_wt_k<<<blks(208 * Edim), 256, 0, stream>>>(post_W1 + (size_t)200 * 200, 200, Edim, 200,
                                                    postW1eT, Edim, 208 * Edim);
    prep_wt_k<<<blks(1024 * HPAD), 256, 0, stream>>>(post_W2, 1024, 200, 1024, postW2T, HPAD, 1024 * HPAD);

    init_bf_k<<<blks(Bsz * LCdim), 256, 0, stream>>>(stoch0, s_bf, LCdim, LCdim, Bsz * LCdim);
    init_bf_k<<<blks(Bsz * HPAD), 256, 0, stream>>>(determ0, h_bf, Hdim, HPAD, Bsz * HPAD);
    copy_f_k<<<blks(Bsz * Hdim), 256, 0, stream>>>(determ0, h_f32, Bsz * Hdim);
    zero_us_k<<<blks(Bsz * HPAD), 256, 0, stream>>>(x_bf, Bsz * HPAD);
    zero_us_k<<<blks(Bsz * HPAD), 256, 0, stream>>>(eh_bf, Bsz * HPAD);
    zero_us_k<<<blks(Bsz * HPAD), 256, 0, stream>>>(ph_bf, Bsz * HPAD);

    // -------------------------- the sequential scan -------------------------
    for (int t = 0; t < Tlen; ++t) {
        const float* emb_t = embed + (size_t)t * Bsz * Edim;
        const float* act_t = action + (size_t)t * Bsz * 6;
        float* out_t = out + (size_t)t * Bsz * OUTC;

        // host-side threefry: ensemble index (fold_in(key42, 2t) -> randint % 5)
        unsigned a0, a1, b0, b1, sk0, sk1;
        tf2x32(0u, 42u, 0u, (unsigned)(2 * t), a0, a1);
        tf2x32(a0, a1, 0u, 0u, b0, b1);
        int eidx = (int)(b0 % 5u);
        tf2x32(0u, 42u, 0u, (unsigned)(2 * t + 1), sk0, sk1);  // sampling key

        conv_bf_k<<<blks(Bsz * Edim), 256, 0, stream>>>(emb_t, embT, Bsz * Edim);

        // x = elu(ln([s,a] @ pre_W + b)); action part as fp32 tail (K=6)
        gemm_ln_elu_k<<<dim3(Bsz / 16), 128, 0, stream>>>(
            s_bf, LCdim, preWt, 1024, 32,
            nullptr, 0, nullptr, 0, 0,
            pre_b, pre_ln_g, pre_ln_b,
            act_t, 6, pre_W + (size_t)1024 * 200, 200, 6,
            x_bf, HPAD, 13, Hdim);

        // GRU gi/gh
        gemm_plain_k<<<dim3(Bsz / 16, 10), 128, 0, stream>>>(
            x_bf, HPAD, WihT, HPAD, 7, gru_bih, gi, 600, 600, 38);
        gemm_plain_k<<<dim3(Bsz / 16, 10), 128, 0, stream>>>(
            h_bf, HPAD, WhhT, HPAD, 7, gru_bhh, gh, 600, 600, 38);
        gru_gate_k<<<blks(Bsz * Hdim), 256, 0, stream>>>(gi, gh, h_f32, h_bf, out_t);

        // selected ensemble member only (identical to reference's gather)
        gemm_ln_elu_k<<<dim3(Bsz / 16), 128, 0, stream>>>(
            h_bf, HPAD, ensW1T + (size_t)eidx * 208 * HPAD, HPAD, 7,
            nullptr, 0, nullptr, 0, 0,
            ens_b1 + (size_t)eidx * 200, ens_ln_g + (size_t)eidx * 200,
            ens_ln_b + (size_t)eidx * 200,
            nullptr, 0, nullptr, 0, 0,
            eh_bf, HPAD, 13, Hdim);
        gemm_plain_k<<<dim3(Bsz / 16, 16), 128, 0, stream>>>(
            eh_bf, HPAD, ensW2T + (size_t)eidx * 1024 * HPAD, HPAD, 7,
            ens_b2 + (size_t)eidx * 1024, out_t + 200, OUTC, 1024, 64);

        // posterior: [h_new, emb] @ post_W1 -> LN -> ELU ; then @ post_W2
        gemm_ln_elu_k<<<dim3(Bsz / 16), 128, 0, stream>>>(
            h_bf, HPAD, postW1hT, HPAD, 7,
            embT, Edim, postW1eT, Edim, 48,
            post_b1, post_ln_g, post_ln_b,
            nullptr, 0, nullptr, 0, 0,
            ph_bf, HPAD, 13, Hdim);
        gemm_plain_k<<<dim3(Bsz / 16, 16), 128, 0, stream>>>(
            ph_bf, HPAD, postW2T, HPAD, 7, post_b2, out_t + 1224, OUTC, 1024, 64);

        // categorical straight-through sample -> s_new (d_out) + s_bf (carry)
        sample_k<<<dim3(Bsz), 256, 0, stream>>>(out_t + 1224, out_t + 2248, s_bf, sk0, sk1);
    }
}
